// CausalAttention_62285615727326
// MI455X (gfx1250) — matrix operand order repaired and checked by reading, not compiled, not run on hardware
//
#include <hip/hip_runtime.h>
#include <hip/hip_bf16.h>
#include <math.h>

// ---------------------------------------------------------------------------
// CausalAttention (reference mask is all-false -> plain full softmax)
//   x [4,4096,512] f32, w_q/w_k/w_v [512,512] f32 -> out [4,4096,512] f32
// bf16 WMMA (v_wmma_f32_16x16x32_bf16) + flash-style online softmax.
// Attention K/V staged through LDS with the CDNA5 Tensor Data Mover
// (tensor_load_to_lds, TENSORcnt) and shared by all 4 waves of a block.
// ---------------------------------------------------------------------------

#define BATCH 4
#define SEQ   4096
#define DIM   512

typedef __attribute__((ext_vector_type(16))) __bf16 v16bf;
typedef __attribute__((ext_vector_type(8)))  __bf16 v8bf;
typedef __attribute__((ext_vector_type(8)))  float  v8f;
typedef unsigned int v4u __attribute__((ext_vector_type(4)));
typedef unsigned int v8u __attribute__((ext_vector_type(8)));

// LDS layout of the attention kernel (dynamic shared memory, bytes):
//   sK : 2 buffers x 32 rows x 520 bf16  (512 data + 4-dword TDM pad) = 66560
//   sV : 2 buffers x 512 rows x 40 bf16  ( 32 data + 4-dword TDM pad) = 81920
//   sP : 4 waves x 16 x 32 f32 (P transpose scratch)                  =  8192
#define K_ROW_ELEMS 520
#define V_ROW_ELEMS 40
#define K_BUF_ELEMS (32 * K_ROW_ELEMS)
#define V_BUF_ELEMS (512 * V_ROW_ELEMS)
#define SK_BYTES (2 * K_BUF_ELEMS * 2)
#define SV_BYTES (2 * V_BUF_ELEMS * 2)
#define SP_BYTES (4 * 16 * 32 * 4)
#define ATTN_LDS_BYTES (SK_BYTES + SV_BYTES + SP_BYTES)   // 156672

// --- WMMA wrapper ----------------------------------------------------------
__device__ __forceinline__ v8f wmma_bf16(v16bf a, v16bf b, v8f c) {
  return __builtin_amdgcn_wmma_f32_16x16x32_bf16(
      false, a, false, b, (short)0, c, false, false);
}

// --- A operand: 16x32 bf16 tile, row-major source, given element stride.
// lanes 0-15: row M=lane, K 0..7 & 16..23 ; lanes 16-31: row M=lane-16, K 8..15 & 24..31
__device__ __forceinline__ v16bf load_a16x32(const __bf16* base, int stride) {
  int lane = threadIdx.x & 31;
  int row  = lane & 15;
  int ko   = (lane >> 4) * 8;
  const __bf16* p = base + (size_t)row * stride + ko;
  v8bf lo = *(const v8bf*)(p);
  v8bf hi = *(const v8bf*)(p + 16);
  v16bf r;
#pragma unroll
  for (int i = 0; i < 8; ++i) { r[i] = lo[i]; r[8 + i] = hi[i]; }
  return r;
}

// --- B operand: 32x16 bf16 tile. Column n in lane n; lanes 0-15 K 0..7 & 16..23,
// lanes 16-31 K 8..15 & 24..31. Source row index = B column, contiguous dim = K.
__device__ __forceinline__ v16bf load_b32x16(const __bf16* base, int stride) {
  int lane = threadIdx.x & 31;
  int col  = lane & 15;
  int ko   = (lane >> 4) * 8;
  const __bf16* p = base + (size_t)col * stride + ko;
  v8bf lo = *(const v8bf*)(p);
  v8bf hi = *(const v8bf*)(p + 16);
  v16bf r;
#pragma unroll
  for (int i = 0; i < 8; ++i) { r[i] = lo[i]; r[8 + i] = hi[i]; }
  return r;
}

// --- Tensor Data Mover: 2D tile global->LDS (D# groups 0+1 only, 2D form) --
// tileW elements per row, tileH rows, strideElems between rows,
// padIntervalCode: LDS pad fires every 2^(code+1) dwords; padAmountCode: (code+1) dwords.
__device__ __forceinline__ void tdm_load_2d(unsigned ldsOff, const void* gptr,
                                            unsigned tileW, unsigned tileH,
                                            unsigned strideElems,
                                            unsigned padIntervalCode,
                                            unsigned padAmountCode) {
  unsigned long long ga = (unsigned long long)(uintptr_t)gptr;
  v4u g0;
  g0[0] = 1u;                                        // count=1, user descriptor
  g0[1] = ldsOff;                                    // lds_addr (bytes)
  g0[2] = (unsigned)(ga & 0xffffffffu);              // global_addr[31:0]
  g0[3] = (unsigned)((ga >> 32) & 0x01ffffffu)       // global_addr[56:32]
        | (2u << 30);                                // type = 2 ("image")
  v8u g1;
  g1[0] = (1u << 16)                                 // data_size = 2 bytes
        | (1u << 20)                                 // pad_enable
        | (padIntervalCode << 22)
        | (padAmountCode << 25);
  g1[1] = (tileW & 0xffffu) << 16;                   // tensor_dim0[15:0]
  g1[2] = (tileW >> 16) | ((tileH & 0xffffu) << 16); // tensor_dim0[31:16] | tensor_dim1[15:0]
  g1[3] = (tileH >> 16) | (tileW << 16);             // tensor_dim1[31:16] | tile_dim0
  g1[4] = tileH;                                     // tile_dim1 (tile_dim2 = 0)
  g1[5] = strideElems;                               // tensor_dim0_stride[31:0]
  g1[6] = 0u;
  g1[7] = 0u;
  asm volatile("tensor_load_to_lds %0, %1" :: "s"(g0), "s"(g1) : "memory");
}

__device__ __forceinline__ void wait_tensorcnt0() {
#if __has_builtin(__builtin_amdgcn_s_wait_tensorcnt)
  __builtin_amdgcn_s_wait_tensorcnt(0);
#else
  asm volatile("s_wait_tensorcnt 0x0" ::: "memory");
#endif
}
__device__ __forceinline__ void wait_tensorcnt2() {
#if __has_builtin(__builtin_amdgcn_s_wait_tensorcnt)
  __builtin_amdgcn_s_wait_tensorcnt(2);
#else
  asm volatile("s_wait_tensorcnt 0x2" ::: "memory");
#endif
}

// ---------------------------------------------------------------------------
// Kernel 0: f32 -> bf16 conversion
// ---------------------------------------------------------------------------
__global__ void cvt_bf16_kernel(const float* __restrict__ src,
                                __bf16* __restrict__ dst, int n) {
  int i = blockIdx.x * blockDim.x + threadIdx.x;
  if (i < n) dst[i] = (__bf16)src[i];
}

// ---------------------------------------------------------------------------
// Kernel 1: projections q = x wq^T, k = x wk^T, v = x wv^T (one wave / tile)
// outputs: qb, kb row-major [BATCH*SEQ, DIM]; vTb transposed [BATCH][DIM][SEQ]
// ---------------------------------------------------------------------------
__global__ void proj_kernel(const __bf16* __restrict__ xb,
                            const __bf16* __restrict__ wqb,
                            const __bf16* __restrict__ wkb,
                            const __bf16* __restrict__ wvb,
                            __bf16* __restrict__ qb,
                            __bf16* __restrict__ kb,
                            __bf16* __restrict__ vTb) {
  const int rowTiles = (BATCH * SEQ) / 16;   // 1024
  const int oTiles   = DIM / 16;             // 32
  int wave = blockIdx.x * (blockDim.x >> 5) + (threadIdx.x >> 5);
  int oT   = wave % oTiles;
  int tmp  = wave / oTiles;
  int rT   = tmp % rowTiles;
  int mat  = tmp / rowTiles;                 // 0=q 1=k 2=v

  const __bf16* w = (mat == 0) ? wqb : (mat == 1) ? wkb : wvb;
  int row0 = rT * 16;
  int o0   = oT * 16;

  v8f acc = {};
#pragma unroll
  for (int dc = 0; dc < DIM / 32; ++dc) {
    v16bf a = load_a16x32(xb + (size_t)row0 * DIM + dc * 32, DIM);
    v16bf b = load_b32x16(w + (size_t)o0 * DIM + dc * 32, DIM);
    acc = wmma_bf16(a, b, acc);
  }

  int lane = threadIdx.x & 31;
  int c = lane & 15, hi = lane >> 4;
  if (mat < 2) {
    __bf16* dst = (mat == 0) ? qb : kb;
#pragma unroll
    for (int r = 0; r < 8; ++r) {
      int m = r + 8 * hi;
      dst[(size_t)(row0 + m) * DIM + o0 + c] = (__bf16)acc[r];
    }
  } else {
    int bidx = row0 / SEQ;
    int n0   = row0 % SEQ;
    v8bf pk;
#pragma unroll
    for (int r = 0; r < 8; ++r) pk[r] = (__bf16)acc[r];
    __bf16* dst = vTb + ((size_t)bidx * DIM + o0 + c) * SEQ + n0 + 8 * hi;
    *(v8bf*)dst = pk;
  }
}

// ---------------------------------------------------------------------------
// Kernel 2: flash attention. 4 waves/block, each owns a 16-query tile of the
// same batch; K and V^T tiles staged via TDM into double-buffered LDS and
// shared by all waves. Q lives in VGPRs; O accumulator [16x512] f32 in VGPRs.
// ---------------------------------------------------------------------------
__global__ void __launch_bounds__(128, 1)
attn_kernel(const __bf16* __restrict__ qb,
            const __bf16* __restrict__ kbm,
            const __bf16* __restrict__ vTb,
            float* __restrict__ out) {
  extern __shared__ char smem[];
  __bf16* sK = (__bf16*)smem;                          // [2][32][520]
  __bf16* sV = (__bf16*)(smem + SK_BYTES);             // [2][512][40]
  float*  sP = (float*)(smem + SK_BYTES + SV_BYTES);   // [4][16][32]

  int wv   = threadIdx.x >> 5;
  int bidx = blockIdx.x / 64;                   // 4 batches, 64 blocks each
  int qT   = (blockIdx.x % 64) * 4 + wv;        // 256 q-tiles per batch
  const __bf16* qB = qb  + (size_t)bidx * SEQ * DIM;
  const __bf16* kB = kbm + (size_t)bidx * SEQ * DIM;
  const __bf16* vB = vTb + (size_t)bidx * DIM * SEQ;
  int n0   = qT * 16;
  int lane = threadIdx.x & 31;
  int c = lane & 15, hi = lane >> 4;
  bool leader = (threadIdx.x < 32);

  // ---- Q block in registers: 16 A-tiles (16x32 each) = 128 VGPRs ----
  v16bf Qa[DIM / 32];
#pragma unroll
  for (int dc = 0; dc < DIM / 32; ++dc)
    Qa[dc] = load_a16x32(qB + (size_t)n0 * DIM + dc * 32, DIM);

  v8f O[DIM / 16];
#pragma unroll
  for (int t = 0; t < DIM / 16; ++t) O[t] = (v8f){};
  float m_run[8], l_run[8];
#pragma unroll
  for (int r = 0; r < 8; ++r) { m_run[r] = -__builtin_inff(); l_run[r] = 0.f; }
  const float scale = 0.044194173824159216f;    // 1/sqrt(512)

  // ---- prologue: TDM-prefetch first K/V tiles into buffer 0 ----
  if (leader) {
    // K tile: 32 rows x 512 bf16, pad every 256 dwords (row) by 4 dwords
    tdm_load_2d((unsigned)(uintptr_t)sK, kB, DIM, 32, DIM, 7, 3);
    // V tile: 512 rows x 32 bf16 (from vT), pad every 16 dwords by 4 dwords
    tdm_load_2d((unsigned)(uintptr_t)sV, vB, 32, DIM, SEQ, 3, 3);
  }

  int cur = 0;
  for (int k0 = 0; k0 < SEQ; k0 += 32) {
    if (leader) {
      if (k0 + 32 < SEQ) {
        int nxt = cur ^ 1;
        tdm_load_2d((unsigned)(uintptr_t)(sK + nxt * K_BUF_ELEMS),
                    kB + (size_t)(k0 + 32) * DIM, DIM, 32, DIM, 7, 3);
        tdm_load_2d((unsigned)(uintptr_t)(sV + nxt * V_BUF_ELEMS),
                    vB + (k0 + 32), 32, DIM, SEQ, 3, 3);
        wait_tensorcnt2();      // the 2 just-issued may fly; 'cur' pair is done
      } else {
        wait_tensorcnt0();
      }
    }
    __syncthreads();            // publish buffer 'cur' to all 4 waves

    const __bf16* bK = sK + cur * K_BUF_ELEMS;
    const __bf16* bV = sV + cur * V_BUF_ELEMS;

    // ---- S = Q * K^T for 32 keys (two 16-col C tiles), K from LDS ----
    v8f S0 = {}, S1 = {};
#pragma unroll
    for (int dc = 0; dc < DIM / 32; ++dc) {
      v16bf b0 = load_b32x16(bK + dc * 32, K_ROW_ELEMS);
      v16bf b1 = load_b32x16(bK + 16 * K_ROW_ELEMS + dc * 32, K_ROW_ELEMS);
      S0 = wmma_bf16(Qa[dc], b0, S0);
      S1 = wmma_bf16(Qa[dc], b1, S1);
    }

    // ---- online softmax (row = one VGPR across a 16-lane half) ----
    float al[8];
#pragma unroll
    for (int r = 0; r < 8; ++r) {
      S0[r] *= scale; S1[r] *= scale;
      float t = fmaxf(S0[r], S1[r]);
      t = fmaxf(t, __shfl_xor(t, 1));
      t = fmaxf(t, __shfl_xor(t, 2));
      t = fmaxf(t, __shfl_xor(t, 4));
      t = fmaxf(t, __shfl_xor(t, 8));
      float mn = fmaxf(m_run[r], t);
      al[r] = __expf(m_run[r] - mn);
      float p0 = __expf(S0[r] - mn);
      float p1 = __expf(S1[r] - mn);
      S0[r] = p0; S1[r] = p1;
      float rs = p0 + p1;
      rs += __shfl_xor(rs, 1);
      rs += __shfl_xor(rs, 2);
      rs += __shfl_xor(rs, 4);
      rs += __shfl_xor(rs, 8);
      l_run[r] = l_run[r] * al[r] + rs;
      m_run[r] = mn;
    }
#pragma unroll
    for (int t = 0; t < DIM / 16; ++t)
#pragma unroll
      for (int r = 0; r < 8; ++r) O[t][r] *= al[r];

    // ---- C-layout P -> A-layout bf16 via wave-private LDS scratch ----
    float* myP = sP + wv * 16 * 32;
#pragma unroll
    for (int r = 0; r < 8; ++r) {
      int q = r + 8 * hi;
      myP[q * 32 + c]      = S0[r];
      myP[q * 32 + 16 + c] = S1[r];
    }
    asm volatile("s_wait_dscnt 0x0" ::: "memory");
    v16bf pa;
    {
      int q  = lane & 15;
      int ko = hi * 8;
#pragma unroll
      for (int i = 0; i < 8; ++i) {
        pa[i]     = (__bf16)myP[q * 32 + ko + i];
        pa[8 + i] = (__bf16)myP[q * 32 + 16 + ko + i];
      }
    }

    // ---- O += P * V, V columns from padded LDS slab of v^T ----
#pragma unroll
    for (int t = 0; t < DIM / 16; ++t) {
      v16bf b = load_b32x16(bV + (t * 16) * V_ROW_ELEMS, V_ROW_ELEMS);
      O[t] = wmma_bf16(pa, b, O[t]);
    }

    __syncthreads();            // all waves done with 'cur' before next prefetch
    cur ^= 1;
  }

  // ---- epilogue ----
  float linv[8];
#pragma unroll
  for (int r = 0; r < 8; ++r) linv[r] = 1.0f / l_run[r];
  float* oB = out + ((size_t)bidx * SEQ + n0) * DIM;
#pragma unroll
  for (int t = 0; t < DIM / 16; ++t)
#pragma unroll
    for (int r = 0; r < 8; ++r) {
      int m = r + 8 * hi;
      oB[(size_t)m * DIM + t * 16 + c] = O[t][r] * linv[r];
    }
}

// ---------------------------------------------------------------------------
extern "C" void kernel_launch(void* const* d_in, const int* in_sizes, int n_in,
                              void* d_out, int out_size, void* d_ws, size_t ws_size,
                              hipStream_t stream) {
  const float* x   = (const float*)d_in[0];
  const float* w_q = (const float*)d_in[1];
  const float* w_k = (const float*)d_in[2];
  const float* w_v = (const float*)d_in[3];
  float* out = (float*)d_out;

  const size_t nx = (size_t)BATCH * SEQ * DIM;  // 8388608
  const size_t nw = (size_t)DIM * DIM;          // 262144

  // workspace (bf16): xb | wq | wk | wv | q | k | vT   (~66 MB)
  __bf16* xb  = (__bf16*)d_ws;
  __bf16* wqb = xb  + nx;
  __bf16* wkb = wqb + nw;
  __bf16* wvb = wkb + nw;
  __bf16* qbp = wvb + nw;
  __bf16* kbp = qbp + nx;
  __bf16* vTp = kbp + nx;

  const int thr = 256;
  cvt_bf16_kernel<<<(int)((nx + thr - 1) / thr), thr, 0, stream>>>(x, xb, (int)nx);
  cvt_bf16_kernel<<<(int)((nw + thr - 1) / thr), thr, 0, stream>>>(w_q, wqb, (int)nw);
  cvt_bf16_kernel<<<(int)((nw + thr - 1) / thr), thr, 0, stream>>>(w_k, wkb, (int)nw);
  cvt_bf16_kernel<<<(int)((nw + thr - 1) / thr), thr, 0, stream>>>(w_v, wvb, (int)nw);

  // projections: 3 * 1024 * 32 = 98304 waves, 8 waves/block
  proj_kernel<<<98304 / 8, 256, 0, stream>>>(xb, wqb, wkb, wvb, qbp, kbp, vTp);

  // attention: 1024 waves, 4 waves/block sharing K/V via TDM-staged LDS
  attn_kernel<<<1024 / 4, 128, ATTN_LDS_BYTES, stream>>>(qbp, kbp, vTp, out);
}
